// Attacker_optimization_90417651516031
// MI455X (gfx1250) — compile-verified, matrix operand order checked by reading, not run on hardware
//
#include <hip/hip_runtime.h>

#define NN     8192
#define TILE   64
#define PITCH  65          // padded LDS row pitch (bank-conflict-friendly)
#define GROUPS 8           // gridDim.x: column groups
#define TPG    16          // column tiles per group (8*16*64 = 8192)
#define TELE   0.15f       // PageRank teleport constant

typedef __attribute__((ext_vector_type(16))) _Float16 v16h;
typedef __attribute__((ext_vector_type(8)))  float    v8f;

// Bsym element for local row u / local col k of the current tile.
// MODE 0: bi < bj  -> Bsym = B[i][j]  = Bl[u][k]
// MODE 1: bi > bj  -> Bsym = B[j][i]  = Bl[k][u]   (transposed LDS read)
// MODE 2: diagonal -> per-element select, zero diagonal
template<int MODE>
__device__ __forceinline__ float bsym_at(const float* __restrict__ Bl, int u, int k) {
  if (MODE == 0) return Bl[u * PITCH + k];
  if (MODE == 1) return Bl[k * PITCH + u];
  return (k > u) ? Bl[u * PITCH + k] : ((k < u) ? Bl[k * PITCH + u] : 0.0f);
}

// One 16x32 P-fragment (optionally scaled by t[col]) reduced along K via
// WMMA with an all-ones B operand: every column of D accumulates row sums.
template<int MODE, bool PASS2>
__device__ __forceinline__ v8f tile_mma(const float* __restrict__ Bl,
                                        const float* __restrict__ Sl,
                                        const float* __restrict__ Tl,
                                        int u, int kbase, int half, v8f c)
{
  v16h a;
#pragma unroll
  for (int e = 0; e < 8; ++e) {
    const int k1 = kbase + half * 8 + e;   // K in [kbase, kbase+16)
    const int k2 = k1 + 16;                // K in [kbase+16, kbase+32)
    float d1 = bsym_at<MODE>(Bl, u, k1) - Sl[u * PITCH + k1];
    float d2 = bsym_at<MODE>(Bl, u, k2) - Sl[u * PITCH + k2];
    // Zero-cost register barrier: keeps the sub/fabs chain scalar so the SLP
    // vectorizer can't form the packed-f32 fabs pattern that crashes the
    // gfx1250 instruction selector (v2f32 bitcast/and ISel bug).
    asm("" : "+v"(d1), "+v"(d2));
    float p1 = __builtin_fabsf(d1);
    float p2 = __builtin_fabsf(d2);
    if (PASS2) { p1 *= Tl[k1]; p2 *= Tl[k2]; }
    a[e]     = (_Float16)p1;
    a[e + 8] = (_Float16)p2;
  }
  v16h ones;
#pragma unroll
  for (int e = 0; e < 16; ++e) ones[e] = (_Float16)1.0f;
  // D = A(16x32,f16) * ones(32x16,f16) + C(16x16,f32)
  return __builtin_amdgcn_wmma_f32_16x16x32_f16(false, a, false, ones,
                                                (short)0, c, false, false);
}

// One pass over the implicit matrix.
// PASS2=false: partial[g][i] = partial row sums of P            (for rs)
// PASS2=true : partial[g][i] = partial row sums of P .* t[col]  (for out)
template<bool PASS2>
__global__ __launch_bounds__(256)
void pr_pass(const float* __restrict__ B, const float* __restrict__ S,
             const float* __restrict__ tvec, float* __restrict__ partial)
{
  __shared__ float Bl[TILE * PITCH];
  __shared__ float Sl[TILE * PITCH];
  __shared__ float Tl[TILE];
  __shared__ float rowacc[TILE];

  const int tid   = threadIdx.x;
  const int lane  = tid & 31;
  const int wave  = tid >> 5;
  const int band  = wave & 3;        // 16-row band within 64-row tile
  const int khalf = wave >> 2;       // which 32-column half of the tile
  const int m     = lane & 15;       // WMMA A: lane&15 == M (row in band)
  const int half  = lane >> 4;
  const int u     = band * 16 + m;   // local row 0..63
  const int kbase = khalf * 32;

  const int bi = blockIdx.y;
  const int r0 = bi * TILE;

  v8f c = {};

  for (int it = 0; it < TPG; ++it) {
    const int bj = blockIdx.x * TPG + it;
    const int c0 = bj * TILE;
    // Bsym only ever reads B's upper-triangle source block (min,max):
    const int bgr = (bi <= bj) ? r0 : c0;
    const int bgc = (bi <= bj) ? c0 : r0;

    __syncthreads();                 // previous iteration's compute done
#pragma unroll
    for (int i = 0; i < 4; ++i) {    // 256 thr * 4 float4 = 64x64 tile
      const int idx = tid + i * 256;
      const int r = idx >> 4;
      const int q = (idx & 15) << 2;
      const float4 bv = *(const float4*)(B + (size_t)(bgr + r) * NN + (bgc + q));
      const float4 sv = *(const float4*)(S + (size_t)(r0 + r) * NN + (c0 + q));
      float* bd = &Bl[r * PITCH + q];
      bd[0] = bv.x; bd[1] = bv.y; bd[2] = bv.z; bd[3] = bv.w;
      float* sd = &Sl[r * PITCH + q];
      sd[0] = sv.x; sd[1] = sv.y; sd[2] = sv.z; sd[3] = sv.w;
    }
    if (PASS2 && tid < TILE) Tl[tid] = tvec[c0 + tid];

    if (it + 1 < TPG) {              // prefetch next tile (global_prefetch_b8)
      const int bj2 = bj + 1;
      const int c02 = c0 + TILE;
      const int bgr2 = (bi <= bj2) ? r0 : c02;
      const int bgc2 = (bi <= bj2) ? c02 : r0;
      const int r = tid >> 4;
      const int q = (tid & 15) << 2;
      __builtin_prefetch(B + (size_t)(bgr2 + r) * NN + (bgc2 + q), 0, 1);
      __builtin_prefetch(S + (size_t)(r0 + r) * NN + (c02 + q), 0, 1);
    }
    __syncthreads();

    if (bi < bj)      c = tile_mma<0, PASS2>(Bl, Sl, Tl, u, kbase, half, c);
    else if (bi > bj) c = tile_mma<1, PASS2>(Bl, Sl, Tl, u, kbase, half, c);
    else              c = tile_mma<2, PASS2>(Bl, Sl, Tl, u, kbase, half, c);
  }

  // Column N=0 of the 16x16 f32 accumulator holds the row sums.
  // C/D layout: VGPR r, lane L -> M = r + 8*(L>=16), N = L&15.
  __syncthreads();
  if (m == 0 && khalf == 0) {
#pragma unroll
    for (int r = 0; r < 8; ++r) rowacc[band * 16 + half * 8 + r] = c[r];
  }
  __syncthreads();
  if (m == 0 && khalf == 1) {
#pragma unroll
    for (int r = 0; r < 8; ++r) rowacc[band * 16 + half * 8 + r] += c[r];
  }
  __syncthreads();
  if (tid < TILE)
    partial[(size_t)blockIdx.x * NN + (r0 + tid)] = rowacc[tid];
}

// t[j] = (1-C) * s[j] / rowsum(j), rowsum reduced in fixed order (deterministic)
__global__ void pr_mid(const float* __restrict__ partial,
                       const float* __restrict__ s, float* __restrict__ t)
{
  const int i = blockIdx.x * blockDim.x + threadIdx.x;
  if (i < NN) {
    float rs = 0.0f;
#pragma unroll
    for (int g = 0; g < GROUPS; ++g) rs += partial[(size_t)g * NN + i];
    t[i] = (1.0f - TELE) * s[i] / rs;
  }
}

// out[i] = C/n + sum_g partial2[g][i]   ((1-C) already folded into t)
__global__ void pr_final(const float* __restrict__ partial, float* __restrict__ out)
{
  const int i = blockIdx.x * blockDim.x + threadIdx.x;
  if (i < NN) {
    float acc = TELE / (float)NN;
#pragma unroll
    for (int g = 0; g < GROUPS; ++g) acc += partial[(size_t)g * NN + i];
    out[i] = acc;
  }
}

extern "C" void kernel_launch(void* const* d_in, const int* in_sizes, int n_in,
                              void* d_out, int out_size, void* d_ws, size_t ws_size,
                              hipStream_t stream)
{
  const float* B = (const float*)d_in[0];          // [8192, 8192] f32
  const float* S = (const float*)d_in[1];          // [8192, 8192] f32
  const float* s = (const float*)d_in[2];          // [8192] f32
  float* out     = (float*)d_out;                  // [8192] f32

  float* partial = (float*)d_ws;                   // GROUPS*NN f32 (256 KB)
  float* t       = partial + (size_t)GROUPS * NN;  // NN f32

  dim3 grid(GROUPS, NN / TILE);                    // (8, 128)

  pr_pass<false><<<grid, 256, 0, stream>>>(B, S, t, partial);  // t unused here
  pr_mid<<<NN / 256, 256, 0, stream>>>(partial, s, t);
  pr_pass<true><<<grid, 256, 0, stream>>>(B, S, t, partial);
  pr_final<<<NN / 256, 256, 0, stream>>>(partial, out);
}
